// FlashAttention_28020366639541
// MI455X (gfx1250) — compile-verified, matrix operand order checked by reading, not run on hardware
//
#include <hip/hip_runtime.h>
#include <hip/hip_bf16.h>
#include <stdint.h>

// ---------------- CDNA5 WMMA types ----------------
typedef __attribute__((ext_vector_type(16))) __bf16 v16bf;
typedef __attribute__((ext_vector_type(8)))  float  v8f;
typedef __attribute__((ext_vector_type(4)))  int    v4i;

#define EMBED 1024
#define HEADS 16
#define HDIM  64
#define BATCH 2
#define SEQ   2048
#define ROWS  (BATCH * SEQ)   // 4096

// log2(e) / sqrt(Dh): folded into the Q projection so attention scores come
// out of WMMA ready for exp2.
#define QSCALE 0.18033688011112042f

__device__ __forceinline__ unsigned short f2bf(float f) {
    union { float f; unsigned u; } c; c.f = f;
    unsigned r = 0x7FFFu + ((c.u >> 16) & 1u);   // round-to-nearest-even
    return (unsigned short)((c.u + r) >> 16);
}

// 32-byte fragment viewed as 2x uint4 (for b128 loads) or v16bf (for WMMA)
union Frag {
    uint4 q[2];
    v16bf v;
    unsigned short s[16];
};

__device__ __forceinline__ v8f bfwmma(const Frag& a, const Frag& b, v8f c) {
    return __builtin_amdgcn_wmma_f32_16x16x32_bf16(false, a.v, false, b.v,
                                                   (short)0, c, false, false);
}

// ---------------- async global->LDS staging (CDNA5), sync fallback ----------------
#if defined(__HIP_DEVICE_COMPILE__) && \
    __has_builtin(__builtin_amdgcn_global_load_async_to_lds_b128) && \
    __has_builtin(__builtin_amdgcn_s_wait_asynccnt)
#define USE_ASYNC_LDS 1
#else
#define USE_ASYNC_LDS 0
#endif

#if USE_ASYNC_LDS
typedef __attribute__((address_space(1))) v4i v4i_g;   // global int4
typedef __attribute__((address_space(3))) v4i v4i_l;   // LDS int4
#endif

__device__ __forceinline__ void stage_b128(const unsigned short* g, unsigned short* l) {
#if USE_ASYNC_LDS
    // AS1 keeps the full 64-bit VA; AS3 address is the low 32 bits of the
    // generic LDS address.
    __builtin_amdgcn_global_load_async_to_lds_b128(
        (v4i_g*)(unsigned long long)(uintptr_t)g,
        (v4i_l*)(unsigned int)(uintptr_t)l,
        0, 0);
#else
    *(uint4*)l = *(const uint4*)g;
#endif
}

__device__ __forceinline__ void stage_wait() {
#if USE_ASYNC_LDS
    __builtin_amdgcn_s_wait_asynccnt(0);
#endif
}

// ---------------- fp32 -> bf16 convert ----------------
__global__ void __launch_bounds__(256) cvt_bf16_kernel(const float* __restrict__ src,
                                                       unsigned short* __restrict__ dst,
                                                       int n) {
    int i = (blockIdx.x * 256 + threadIdx.x) * 8;
    if (i + 8 <= n) {
        float4 a = *(const float4*)(src + i);
        float4 b = *(const float4*)(src + i + 4);
        union { unsigned short s[8]; uint4 q; } o;
        o.s[0] = f2bf(a.x); o.s[1] = f2bf(a.y); o.s[2] = f2bf(a.z); o.s[3] = f2bf(a.w);
        o.s[4] = f2bf(b.x); o.s[5] = f2bf(b.y); o.s[6] = f2bf(b.z); o.s[7] = f2bf(b.w);
        *(uint4*)(dst + i) = o.q;
    }
}

// ---------------- tiled bf16 GEMM: C[4096,1024] = A[4096,1024] * B[1024,1024] ----------------
// Block tile 128x64, 8 waves, 32x32 per wave (4 WMMA per K-step, A/B frags reused 2x).
// MODE 0: C = float*, row-major [ROWS, EMBED]
// MODE 2: C = bf16*,  head-split [B, H, S, Dh]
template <int MODE>
__global__ void __launch_bounds__(256) gemm_bf16_kernel(const unsigned short* __restrict__ A,
                                                        const unsigned short* __restrict__ B,
                                                        void* __restrict__ Cout,
                                                        float scale) {
    __shared__ __align__(16) unsigned short As[128][32];   // A tile, row-major
    __shared__ __align__(16) unsigned short Bt[64][32];    // B tile, K-transposed: Bt[n][k]

    const int tid  = threadIdx.x;
    const int lane = tid & 31;
    const int wave = tid >> 5;           // 0..7
    const int wr   = wave >> 1;          // 0..3  (M sub-block, 32 rows each)
    const int wc   = wave & 1;           // 0..1  (N sub-block, 32 cols each)
    const int bm   = blockIdx.y * 128;
    const int bn   = blockIdx.x * 64;

    const int ar = tid >> 1;             // 0..127
    const int ac = (tid & 1) * 16;       // 0,16
    const int br = tid >> 3;             // 0..31
    const int bc = (tid & 7) * 8;        // 0..56

    const int fn = lane & 15;
    const int kb = (lane >> 4) * 8;

    v8f acc[2][2] = {{{}, {}}, {{}, {}}};

    for (int k0 = 0; k0 < EMBED; k0 += 32) {
        const unsigned short* ag = A + (size_t)(bm + ar) * EMBED + k0 + ac;
        stage_b128(ag,     &As[ar][ac]);        // async global->LDS when available
        stage_b128(ag + 8, &As[ar][ac + 8]);

        uint4 bv = *(const uint4*)(B + (size_t)(k0 + br) * EMBED + bn + bc);
        union { uint4 q; unsigned short s[8]; } bu; bu.q = bv;
#pragma unroll
        for (int i = 0; i < 8; ++i) Bt[bc + i][br] = bu.s[i];

        if (k0 + 32 < EMBED)   // global_prefetch_b8 for next B tile
            __builtin_prefetch(B + (size_t)(k0 + 32 + br) * EMBED + bn + bc, 0, 0);

        stage_wait();
        __syncthreads();

        Frag a0, a1, b0, b1;
        a0.q[0] = *(const uint4*)&As[wr * 32 + fn][kb];
        a0.q[1] = *(const uint4*)&As[wr * 32 + fn][kb + 16];
        a1.q[0] = *(const uint4*)&As[wr * 32 + 16 + fn][kb];
        a1.q[1] = *(const uint4*)&As[wr * 32 + 16 + fn][kb + 16];
        b0.q[0] = *(const uint4*)&Bt[wc * 32 + fn][kb];
        b0.q[1] = *(const uint4*)&Bt[wc * 32 + fn][kb + 16];
        b1.q[0] = *(const uint4*)&Bt[wc * 32 + 16 + fn][kb];
        b1.q[1] = *(const uint4*)&Bt[wc * 32 + 16 + fn][kb + 16];

        acc[0][0] = bfwmma(a0, b0, acc[0][0]);
        acc[0][1] = bfwmma(a0, b1, acc[0][1]);
        acc[1][0] = bfwmma(a1, b0, acc[1][0]);
        acc[1][1] = bfwmma(a1, b1, acc[1][1]);
        __syncthreads();
    }

    const int rb = (lane >> 4) * 8;
#pragma unroll
    for (int mi = 0; mi < 2; ++mi) {
#pragma unroll
        for (int ni = 0; ni < 2; ++ni) {
            const int gc = bn + wc * 32 + ni * 16 + fn;
#pragma unroll
            for (int j = 0; j < 8; ++j) {
                int gr = bm + wr * 32 + mi * 16 + rb + j;
                float v = acc[mi][ni][j] * scale;
                if (MODE == 0) {
                    ((float*)Cout)[(size_t)gr * EMBED + gc] = v;
                } else {
                    unsigned short* C = (unsigned short*)Cout;
                    int h = gc >> 6, d = gc & 63;
                    int bb = gr >> 11, ss = gr & 2047;
                    C[((size_t)(bb * HEADS + h) * SEQ + ss) * HDIM + d] = f2bf(v);
                }
            }
        }
    }
}

// ---------------- flash attention (causal, online softmax, exp2 domain) ----------------
// Q pre-scaled by log2(e)/sqrt(Dh). Q/K/V: bf16 [B,H,S,Dh]. Out: bf16 [ROWS, EMBED].
__global__ void __launch_bounds__(256) attn_kernel(const unsigned short* __restrict__ Q,
                                                   const unsigned short* __restrict__ K,
                                                   const unsigned short* __restrict__ V,
                                                   unsigned short* __restrict__ Out) {
    __shared__ __align__(16) unsigned short Ks[64][64];       // K tile row-major [k][d]
    __shared__ __align__(16) unsigned short Vt[64][64];       // V tile transposed [d][k]
    __shared__ __align__(16) unsigned short Pw[8][16][64];    // per-wave P relayout slab

    const int tid  = threadIdx.x;
    const int lane = tid & 31;
    const int wave = tid >> 5;
    const int bh   = blockIdx.y;        // b*16 + h
    const int b    = bh >> 4;
    const int h    = bh & 15;
    const int qb   = blockIdx.x * 128;

    const size_t base = (size_t)bh * SEQ * HDIM;

    const int fn = lane & 15;
    const int kb = (lane >> 4) * 8;
    const int rb = kb;                  // C-layout row base for this half-wave

    Frag qf[2];
    {
        const unsigned short* qrow = Q + base + (size_t)(qb + wave * 16 + fn) * HDIM;
#pragma unroll
        for (int c = 0; c < 2; ++c) {
            qf[c].q[0] = *(const uint4*)(qrow + c * 32 + kb);
            qf[c].q[1] = *(const uint4*)(qrow + c * 32 + kb + 16);
        }
    }

    v8f o0 = {}, o1 = {}, o2 = {}, o3 = {};
    float mrow[8], lrow[8];
#pragma unroll
    for (int j = 0; j < 8; ++j) { mrow[j] = -1e30f; lrow[j] = 0.f; }

    const int lr = tid >> 2;            // 0..63
    const int lc = (tid & 3) * 16;      // 0,16,32,48

    for (int kb0 = 0; kb0 < qb + 128; kb0 += 64) {
        {   // stage K (row-major, async) and V (transposed, via VGPRs) into LDS
            const unsigned short* krow = K + base + (size_t)(kb0 + lr) * HDIM + lc;
            stage_b128(krow,     &Ks[lr][lc]);
            stage_b128(krow + 8, &Ks[lr][lc + 8]);

            const unsigned short* vrow = V + base + (size_t)(kb0 + lr) * HDIM + lc;
            union { uint4 q; unsigned short s[8]; } v0, v1;
            v0.q = *(const uint4*)(vrow);
            v1.q = *(const uint4*)(vrow + 8);
#pragma unroll
            for (int i = 0; i < 8; ++i) {
                Vt[lc + i][lr]     = v0.s[i];
                Vt[lc + 8 + i][lr] = v1.s[i];
            }
            stage_wait();
        }
        __syncthreads();

        const int qlo = qb + wave * 16;
        if (kb0 <= qlo + 15) {          // skip fully-masked key tiles
            // ---- S = Q K^T (16 x 64 strip), fp32 accumulate; scores pre-scaled ----
            v8f st[4];
#pragma unroll
            for (int t = 0; t < 4; ++t) {
                v8f s = {};
#pragma unroll
                for (int c = 0; c < 2; ++c) {
                    Frag bf_;
                    bf_.q[0] = *(const uint4*)&Ks[t * 16 + fn][c * 32 + kb];
                    bf_.q[1] = *(const uint4*)&Ks[t * 16 + fn][c * 32 + kb + 16];
                    s = bfwmma(qf[c], bf_, s);
                }
                st[t] = s;
            }
            // ---- causal mask only on diagonal tiles ----
            if (kb0 + 63 > qlo) {
#pragma unroll
                for (int t = 0; t < 4; ++t) {
                    int kcol = kb0 + t * 16 + fn;
#pragma unroll
                    for (int j = 0; j < 8; ++j) {
                        int qr = qlo + rb + j;
                        st[t][j] = (kcol <= qr) ? st[t][j] : -1e30f;
                    }
                }
            }
            // ---- online softmax (exp2 domain, half-wave shfl_xor reductions) ----
            float mnew[8], alpha[8], rs[8];
#pragma unroll
            for (int j = 0; j < 8; ++j) {
                float rm = fmaxf(fmaxf(st[0][j], st[1][j]), fmaxf(st[2][j], st[3][j]));
#pragma unroll
                for (int d = 1; d < 16; d <<= 1) rm = fmaxf(rm, __shfl_xor(rm, d, 32));
                float mn = fmaxf(mrow[j], rm);
                mnew[j]  = mn;
                alpha[j] = exp2f(mrow[j] - mn);
            }
#pragma unroll
            for (int j = 0; j < 8; ++j) {
                float sum = 0.f;
#pragma unroll
                for (int t = 0; t < 4; ++t) {
                    float p = exp2f(st[t][j] - mnew[j]);
                    st[t][j] = p;
                    sum += p;
                }
#pragma unroll
                for (int d = 1; d < 16; d <<= 1) sum += __shfl_xor(sum, d, 32);
                rs[j] = sum;
            }
#pragma unroll
            for (int j = 0; j < 8; ++j) {
                lrow[j] = alpha[j] * lrow[j] + rs[j];
                mrow[j] = mnew[j];
                o0[j] *= alpha[j]; o1[j] *= alpha[j];
                o2[j] *= alpha[j]; o3[j] *= alpha[j];
            }
            // ---- relayout P (C-layout -> A-layout) via per-wave LDS slab ----
#pragma unroll
            for (int t = 0; t < 4; ++t)
#pragma unroll
                for (int j = 0; j < 8; ++j)
                    Pw[wave][rb + j][t * 16 + fn] = f2bf(st[t][j]);
            Frag pf[2];   // same-wave DS ops are in-order
#pragma unroll
            for (int c = 0; c < 2; ++c) {
                pf[c].q[0] = *(const uint4*)&Pw[wave][fn][c * 32 + kb];
                pf[c].q[1] = *(const uint4*)&Pw[wave][fn][c * 32 + kb + 16];
            }
            // ---- O += P * V ----
#pragma unroll
            for (int t = 0; t < 4; ++t) {
                v8f* op = (t == 0) ? &o0 : (t == 1) ? &o1 : (t == 2) ? &o2 : &o3;
#pragma unroll
                for (int c = 0; c < 2; ++c) {
                    Frag vf;
                    vf.q[0] = *(const uint4*)&Vt[t * 16 + fn][c * 32 + kb];
                    vf.q[1] = *(const uint4*)&Vt[t * 16 + fn][c * 32 + kb + 16];
                    *op = bfwmma(pf[c], vf, *op);
                }
            }
        }
        __syncthreads();
    }

    // ---- finalize: O /= l, write bf16 heads-merged [ROWS, EMBED] ----
#pragma unroll
    for (int j = 0; j < 8; ++j) {
        float inv = 1.0f / lrow[j];
        o0[j] *= inv; o1[j] *= inv; o2[j] *= inv; o3[j] *= inv;
    }
    const int srow = qb + wave * 16 + rb;
    unsigned short* obase = Out + (size_t)b * SEQ * EMBED + (size_t)h * HDIM;
#pragma unroll
    for (int j = 0; j < 8; ++j) {
        unsigned short* orow = obase + (size_t)(srow + j) * EMBED;
        orow[0 * 16 + fn] = f2bf(o0[j]);
        orow[1 * 16 + fn] = f2bf(o1[j]);
        orow[2 * 16 + fn] = f2bf(o2[j]);
        orow[3 * 16 + fn] = f2bf(o3[j]);
    }
}

// ---------------- launcher ----------------
extern "C" void kernel_launch(void* const* d_in, const int* in_sizes, int n_in,
                              void* d_out, int out_size, void* d_ws, size_t ws_size,
                              hipStream_t stream) {
    const float* x  = (const float*)d_in[0];
    const float* Wq = (const float*)d_in[1];
    const float* Wk = (const float*)d_in[2];
    const float* Wv = (const float*)d_in[3];
    const float* Wo = (const float*)d_in[4];
    float* out = (float*)d_out;

    const size_t XE = (size_t)ROWS * EMBED;    // 4,194,304
    const size_t WE = (size_t)EMBED * EMBED;   // 1,048,576

    unsigned short* ws  = (unsigned short*)d_ws;
    unsigned short* xb  = ws; ws += XE;
    unsigned short* wqb = ws; ws += WE;
    unsigned short* wkb = ws; ws += WE;
    unsigned short* wvb = ws; ws += WE;
    unsigned short* wob = ws; ws += WE;
    unsigned short* Qb  = ws; ws += XE;
    unsigned short* Kb  = ws; ws += XE;
    unsigned short* Vb  = ws; ws += XE;
    unsigned short* Ab  = ws; ws += XE;

    cvt_bf16_kernel<<<(int)(XE / 2048), 256, 0, stream>>>(x,  xb,  (int)XE);
    cvt_bf16_kernel<<<(int)(WE / 2048), 256, 0, stream>>>(Wq, wqb, (int)WE);
    cvt_bf16_kernel<<<(int)(WE / 2048), 256, 0, stream>>>(Wk, wkb, (int)WE);
    cvt_bf16_kernel<<<(int)(WE / 2048), 256, 0, stream>>>(Wv, wvb, (int)WE);
    cvt_bf16_kernel<<<(int)(WE / 2048), 256, 0, stream>>>(Wo, wob, (int)WE);

    dim3 gg(EMBED / 64, ROWS / 128);          // (16, 32)
    gemm_bf16_kernel<2><<<gg, 256, 0, stream>>>(xb, wqb, Qb, QSCALE);   // Q pre-scaled
    gemm_bf16_kernel<2><<<gg, 256, 0, stream>>>(xb, wkb, Kb, 1.0f);
    gemm_bf16_kernel<2><<<gg, 256, 0, stream>>>(xb, wvb, Vb, 1.0f);

    dim3 ga(SEQ / 128, BATCH * HEADS);        // (16, 32)
    attn_kernel<<<ga, 256, 0, stream>>>(Qb, Kb, Vb, Ab);

    gemm_bf16_kernel<0><<<gg, 256, 0, stream>>>(Ab, wob, out, 1.0f);
}
